// SimpleVQAutoEncoder_88837103551101
// MI455X (gfx1250) — matrix operand order repaired and checked by reading, not compiled, not run on hardware
//
#include <hip/hip_runtime.h>
#include <hip/hip_bf16.h>
#include <math.h>

typedef __attribute__((ext_vector_type(16))) _Float16     v16h;
typedef __attribute__((ext_vector_type(8)))  float        v8f;
typedef __attribute__((ext_vector_type(4)))  unsigned int u32x4;
typedef __attribute__((ext_vector_type(4)))  int          i32x4;
typedef __attribute__((ext_vector_type(8)))  int          i32x8;

#if __has_builtin(__builtin_amdgcn_tensor_load_to_lds) && __has_builtin(__builtin_amdgcn_s_wait_tensorcnt)
#define USE_TDM 1
#else
#define USE_TDM 0
#endif

__device__ __forceinline__ float gelu_exact(float v) {
    return 0.5f * v * (1.0f + erff(v * 0.70710678118654752f));
}

// ---------------- K0: zero the commit-loss accumulator ----------------
__global__ void k0_zero(float* loss) {
    if (threadIdx.x == 0 && blockIdx.x == 0) *loss = 0.0f;
}

// ---------------- K1: conv1(1->16,3x3,SAME) + maxpool2 + GELU ----------------
__global__ void k1_enc1(const float* __restrict__ x, const float* __restrict__ w1,
                        const float* __restrict__ b1, float* __restrict__ h1) {
    int t  = blockIdx.x * 256 + threadIdx.x;
    int px = t & 127, py = (t >> 7) & 127, b = t >> 14;
    const float* xin = x + b * 65536;
    float p[4][4];
#pragma unroll
    for (int u = 0; u < 4; u++) {
        int iy = 2 * py - 1 + u;
#pragma unroll
        for (int v = 0; v < 4; v++) {
            int ix = 2 * px - 1 + v;
            bool ok = (iy >= 0) && (iy < 256) && (ix >= 0) && (ix < 256);
            p[u][v] = ok ? xin[iy * 256 + ix] : 0.0f;
        }
    }
#pragma unroll
    for (int oc = 0; oc < 16; oc++) {
        float m = -3.4e38f;
#pragma unroll
        for (int q = 0; q < 4; q++) {
            int qy = q >> 1, qx = q & 1;
            float acc = b1[oc];
#pragma unroll
            for (int ky = 0; ky < 3; ky++)
#pragma unroll
                for (int kx = 0; kx < 3; kx++)
                    acc = fmaf(p[qy + ky][qx + kx], w1[oc * 9 + ky * 3 + kx], acc);
            m = fmaxf(m, acc);
        }
        h1[((b * 16 + oc) << 14) + (py << 7) + px] = gelu_exact(m);
    }
}

// ---------------- K2: conv2(16->32,3x3,SAME) + maxpool2 -> f16 tokens ----------------
__global__ void k2_enc2(const float* __restrict__ h1, const float* __restrict__ w2,
                        const float* __restrict__ b2, _Float16* __restrict__ tok) {
    __shared__ float tile[16][18][18];
    int tid = threadIdx.x;
    int tb  = blockIdx.x >> 6;
    int tl  = blockIdx.x & 63;
    int ty0 = (tl >> 3) * 8, tx0 = (tl & 7) * 8;
    int gr0 = 2 * ty0 - 1, gc0 = 2 * tx0 - 1;
    for (int i = tid; i < 16 * 18 * 18; i += 256) {
        int ic = i / 324, r2 = i - ic * 324;
        int ry = r2 / 18, rx = r2 - ry * 18;
        int gy = gr0 + ry, gx = gc0 + rx;
        bool ok = (gy >= 0) && (gy < 128) && (gx >= 0) && (gx < 128);
        tile[ic][ry][rx] = ok ? h1[((tb * 16 + ic) << 14) + (gy << 7) + gx] : 0.0f;
    }
    __syncthreads();
    int ww = tid >> 5;
    int og = ww & 3;                          // uniform within wave -> s_load weights
    int sp = (tid & 31) | ((ww >> 2) << 5);
    int sy = sp >> 3, sx = sp & 7;
    float acc[8][4];
#pragma unroll
    for (int o = 0; o < 8; o++)
#pragma unroll
        for (int q = 0; q < 4; q++) acc[o][q] = 0.0f;
    for (int ic = 0; ic < 16; ic++) {
        float p[4][4];
#pragma unroll
        for (int u = 0; u < 4; u++)
#pragma unroll
            for (int v = 0; v < 4; v++) p[u][v] = tile[ic][2 * sy + u][2 * sx + v];
#pragma unroll
        for (int o = 0; o < 8; o++) {
            int oc = og * 8 + o;
            const float* wp = w2 + (oc * 16 + ic) * 9;
#pragma unroll
            for (int q = 0; q < 4; q++) {
                int qy = q >> 1, qx = q & 1;
                float a = acc[o][q];
#pragma unroll
                for (int ky = 0; ky < 3; ky++)
#pragma unroll
                    for (int kx = 0; kx < 3; kx++)
                        a = fmaf(p[qy + ky][qx + kx], wp[ky * 3 + kx], a);
                acc[o][q] = a;
            }
        }
    }
    int tokid = (tb * 64 + ty0 + sy) * 64 + (tx0 + sx);
#pragma unroll
    for (int o = 0; o < 8; o++) {
        int oc = og * 8 + o;
        float m = fmaxf(fmaxf(acc[o][0], acc[o][1]), fmaxf(acc[o][2], acc[o][3])) + b2[oc];
        tok[tokid * 32 + oc] = (_Float16)m;
    }
}

// ---------------- K3: vector quantize with WMMA + TDM token staging ----------------
// Dynamic-LDS arena (base offset 0 since no static LDS in this kernel):
//   [     0, 32768)  cb   : 512x32 f16 codebook
//   [ 32768, 34816)  cbn  : 512 f32 codebook norms
//   [ 34816, 43008)  st   : 8 waves x 16 tokens x 32 ch f16  (TDM destination)
//   [ 43008, 43520)  nrm  : 8x16 f32 token norms
//   [ 43520, 44032)  idxs : 8x16 i32 chosen indices
#define K3_LDS_CB   0u
#define K3_LDS_CBN  32768u
#define K3_LDS_ST   34816u
#define K3_LDS_NRM  43008u
#define K3_LDS_IDX  43520u
#define K3_LDS_SIZE 44032u

__global__ void k3_vq(const _Float16* __restrict__ tok, const float* __restrict__ cbf,
                      float* __restrict__ hq, int* __restrict__ idx_out,
                      float* __restrict__ loss) {
    extern __shared__ char smem[];
    _Float16 (*cb)[32]      = (_Float16(*)[32])(smem + K3_LDS_CB);
    float*    cbn           = (float*)(smem + K3_LDS_CBN);
    _Float16 (*st)[16][32]  = (_Float16(*)[16][32])(smem + K3_LDS_ST);
    float (*nrm)[16]        = (float(*)[16])(smem + K3_LDS_NRM);
    int   (*idxs)[16]       = (int(*)[16])(smem + K3_LDS_IDX);

    int tid  = threadIdx.x;
    int w    = tid >> 5, lane = tid & 31;
    int tok0 = (blockIdx.x * 8 + w) * 16;

#if USE_TDM
    // One 1-D TDM descriptor per block: 128 tokens x 32 ch f16 = 4096 elements
    // (8 KB contiguous in global) -> LDS st arena. Issued by wave 0 only.
    if (w == 0) {
        unsigned long long ga =
            (unsigned long long)(const void*)(tok + (size_t)blockIdx.x * 4096u);
        // D# group0: count=1 | lds_addr | global_addr[56:0] | type=2
        u32x4 g0 = { 1u, K3_LDS_ST,
                     (unsigned)(ga & 0xFFFFFFFFu),
                     (unsigned)((ga >> 32) & 0x01FFFFFFu) | (2u << 30) };
        // D# group1: data_size=1 (2B); tensor_dim0=4096; tensor_dim1=1;
        //            tile_dim0=4096; tile_dim1=1; tensor_dim0_stride=4096
        i32x8 g1 = { (int)(1u << 16),
                     (int)(4096u << 16),
                     (int)(1u << 16),
                     (int)(4096u << 16),
                     1, 4096, 0, 0 };
        i32x4 gz = { 0, 0, 0, 0 };
#if defined(__clang_major__) && __clang_major__ >= 23
        i32x8 gz8 = { 0, 0, 0, 0, 0, 0, 0, 0 };
        __builtin_amdgcn_tensor_load_to_lds(g0, g1, gz, gz, gz8, 0);
#else
        __builtin_amdgcn_tensor_load_to_lds(g0, g1, gz, gz, 0);
#endif
    }
#else
    {   // manual staging fallback: lane -> token lane>>1, 16-ch half lane&1
        int m = lane >> 1, h = lane & 1;
        const _Float16* src = tok + (tok0 + m) * 32 + h * 16;
#pragma unroll
        for (int j = 0; j < 16; j++) st[w][m][h * 16 + j] = src[j];
    }
#endif

    // stage codebook f32 -> f16 into LDS (overlaps with TDM in flight)
    for (int i = tid; i < 512 * 32; i += 256)
        (&cb[0][0])[i] = (_Float16)cbf[i];
    __builtin_prefetch(cbf + tid * 64, 0, 0);   // warm the f32 gather path

#if USE_TDM
    __builtin_amdgcn_s_wait_tensorcnt(0);       // wave 0's DMA complete
#endif
    __syncthreads();                            // cb + st visible to all

    int g = lane >> 4, nl = lane & 15;

    // per-token squared norms from staged f16 (both halves compute, g==0 writes)
    {
        float s = 0.0f;
#pragma unroll
        for (int j = 0; j < 32; j++) { float f = (float)st[w][nl][j]; s = fmaf(f, f, s); }
        if (g == 0) nrm[w][nl] = s;
    }
    // codebook norms from LDS f16
    for (int c = tid; c < 512; c += 256) {
        float s = 0.0f;
#pragma unroll
        for (int j = 0; j < 32; j++) { float f = (float)cb[c][j]; s = fmaf(f, f, s); }
        cbn[c] = s;
    }
    __syncthreads();

    // A fragment: lane group g, token m = nl,
    // K chunk [16g,16g+16) (contiguous, matches B)
    v16h A;
    {
        const _Float16* tp = &st[w][nl][0];
#pragma unroll
        for (int i = 0; i < 16; i++) A[i] = tp[16 * g + i];
    }

    float best[8]; int bidx[8];
#pragma unroll
    for (int r = 0; r < 8; r++) { best[r] = 3.4e38f; bidx[r] = 0; }

    // 2x-unrolled code-tile loop: two independent WMMA chains per iteration
    for (int t = 0; t < 32; t += 2) {
        v16h B0, B1;
        const _Float16* cp0 = &cb[t * 16 + nl][0];
        const _Float16* cp1 = &cb[t * 16 + 16 + nl][0];
#pragma unroll
        for (int i = 0; i < 16; i++) { B0[i] = cp0[16 * g + i]; B1[i] = cp1[16 * g + i]; }
        v8f C0 = {}, C1 = {};
        C0 = __builtin_amdgcn_wmma_f32_16x16x32_f16(false, A, false, B0,
                                                    (short)0, C0, false, false);
        C1 = __builtin_amdgcn_wmma_f32_16x16x32_f16(false, A, false, B1,
                                                    (short)0, C1, false, false);
        int   n0 = t * 16 + nl,  n1 = n0 + 16;
        float c0 = cbn[n0],      c1 = cbn[n1];
#pragma unroll
        for (int r = 0; r < 8; r++) {          // D layout: VGPR r -> M = r + 8g, N = nl
            float s0 = c0 - 2.0f * C0[r];
            if (s0 < best[r]) { best[r] = s0; bidx[r] = n0; }
            float s1 = c1 - 2.0f * C1[r];
            if (s1 < best[r]) { best[r] = s1; bidx[r] = n1; }
        }
    }

    // argmin reduce across the 16 N-lanes (xor<16 stays within each half-wave)
#pragma unroll
    for (int r = 0; r < 8; r++) {
        float bv = best[r]; int bi = bidx[r];
#pragma unroll
        for (int off = 1; off < 16; off <<= 1) {
            float ob = __shfl_xor(bv, off, 32);
            int   oi = __shfl_xor(bi, off, 32);
            if (ob < bv || (ob == bv && oi < bi)) { bv = ob; bi = oi; }
        }
        best[r] = bv; bidx[r] = bi;
    }

    if (nl == 0) {                 // lane 0 owns tokens 0..7, lane 16 owns 8..15
        float ls = 0.0f;
#pragma unroll
        for (int r = 0; r < 8; r++) {
            int m = r + 8 * g;
            idxs[w][m] = bidx[r];
            idx_out[tok0 + m] = bidx[r];
            ls += best[r] + nrm[w][m];   // = ||f - c||^2 (min distance)
        }
        atomicAdd(loss, ls * (1.0f / (131072.0f * 32.0f)));
    }
    __syncthreads();

    // straight-through forward value: hq = codebook[idx] (f32), scatter to NCHW
    {
        int m  = lane >> 1, h = lane & 1;
        int id = idxs[w][m];
        int tk = tok0 + m;
        int xb = tk & 63, yb = (tk >> 6) & 63, bi = tk >> 12;
        const float* cr  = cbf + id * 32 + h * 16;
        float*       dst = hq + (bi * 32) * 4096 + (yb << 6) + xb;
#pragma unroll
        for (int j = 0; j < 16; j++)
            dst[(h * 16 + j) * 4096] = cr[j];
    }
}

// ---------------- K4: up2 + conv(32->16,3x3,SAME) + GELU ----------------
__global__ void k4_dec1(const float* __restrict__ hq, const float* __restrict__ wd1,
                        const float* __restrict__ bd1, float* __restrict__ y1) {
    int t = blockIdx.x * 256 + threadIdx.x;
    int x = t & 127, y = (t >> 7) & 127, b = t >> 14;
    int rA = (y - 1) >> 1, rB = rA + 1;
    int cA = (x - 1) >> 1, cB = cA + 1;
    bool rAok = (rA >= 0), rBok = (rB < 64);
    bool cAok = (cA >= 0), cBok = (cB < 64);
    float acc[16];
#pragma unroll
    for (int o = 0; o < 16; o++) acc[o] = bd1[o];
    for (int ic = 0; ic < 32; ic++) {
        const float* src = hq + ((b * 32 + ic) << 12);
        __builtin_prefetch(src + 4096, 0, 0);
        float vAA = (rAok && cAok) ? src[(rA << 6) + cA] : 0.0f;
        float vAB = (rAok && cBok) ? src[(rA << 6) + cB] : 0.0f;
        float vBA = (rBok && cAok) ? src[(rB << 6) + cA] : 0.0f;
        float vBB = (rBok && cBok) ? src[(rB << 6) + cB] : 0.0f;
#pragma unroll
        for (int ky = 0; ky < 3; ky++) {
            int sr = ((y - 1 + ky) >> 1) - rA;   // 0 or 1
#pragma unroll
            for (int kx = 0; kx < 3; kx++) {
                int sc = ((x - 1 + kx) >> 1) - cA;
                float u = sr ? (sc ? vBB : vBA) : (sc ? vAB : vAA);
#pragma unroll
                for (int o = 0; o < 16; o++)
                    acc[o] = fmaf(u, wd1[(o * 32 + ic) * 9 + ky * 3 + kx], acc[o]);
            }
        }
    }
#pragma unroll
    for (int o = 0; o < 16; o++)
        y1[((b * 16 + o) << 14) + (y << 7) + x] = gelu_exact(acc[o]);
}

// ---------------- K5: up2 + conv(16->1,3x3,SAME) + clip ----------------
__global__ void k5_dec2(const float* __restrict__ y1, const float* __restrict__ wd2,
                        const float* __restrict__ bd2, float* __restrict__ out) {
    int t = blockIdx.x * 256 + threadIdx.x;
    int x = t & 255, y = (t >> 8) & 255, b = t >> 16;
    int rA = (y - 1) >> 1, rB = rA + 1;
    int cA = (x - 1) >> 1, cB = cA + 1;
    float acc = bd2[0];
    for (int ic = 0; ic < 16; ic++) {
        const float* src = y1 + ((b * 16 + ic) << 14);
        float vAA = (rA >= 0 && cA >= 0)   ? src[(rA << 7) + cA] : 0.0f;
        float vAB = (rA >= 0 && cB < 128)  ? src[(rA << 7) + cB] : 0.0f;
        float vBA = (rB < 128 && cA >= 0)  ? src[(rB << 7) + cA] : 0.0f;
        float vBB = (rB < 128 && cB < 128) ? src[(rB << 7) + cB] : 0.0f;
#pragma unroll
        for (int ky = 0; ky < 3; ky++) {
            int sr = ((y - 1 + ky) >> 1) - rA;
#pragma unroll
            for (int kx = 0; kx < 3; kx++) {
                int sc = ((x - 1 + kx) >> 1) - cA;
                float u = sr ? (sc ? vBB : vBA) : (sc ? vAB : vAA);
                acc = fmaf(u, wd2[ic * 9 + ky * 3 + kx], acc);
            }
        }
    }
    out[b * 65536 + y * 256 + x] = fminf(fmaxf(acc, -1.0f), 1.0f);
}

// ---------------- host launch ----------------
extern "C" void kernel_launch(void* const* d_in, const int* in_sizes, int n_in,
                              void* d_out, int out_size, void* d_ws, size_t ws_size,
                              hipStream_t stream) {
    const float* x   = (const float*)d_in[0];
    const float* e1w = (const float*)d_in[1];
    const float* e1b = (const float*)d_in[2];
    const float* e2w = (const float*)d_in[3];
    const float* e2b = (const float*)d_in[4];
    const float* cbf = (const float*)d_in[5];
    const float* d1w = (const float*)d_in[6];
    const float* d1b = (const float*)d_in[7];
    const float* d2w = (const float*)d_in[8];
    const float* d2b = (const float*)d_in[9];

    float* y       = (float*)d_out;                          // [32,1,256,256]
    int*   idx_out = (int*)((float*)d_out + 2097152);        // [32,64,64]
    float* loss    = (float*)d_out + 2097152 + 131072;       // scalar

    char* ws = (char*)d_ws;
    float*    h1  = (float*)ws;                    // 32*16*128*128 f32 = 33,554,432 B
    _Float16* tok = (_Float16*)(ws + 33554432);    // 131072*32 f16    =  8,388,608 B
    float*    hq  = (float*)(ws + 41943040);       // 32*32*64*64 f32  = 16,777,216 B
    float*    y1  = (float*)(ws + 58720256);       // 32*16*128*128 f32= 33,554,432 B

    k0_zero<<<1, 1, 0, stream>>>(loss);
    k1_enc1<<<2048, 256, 0, stream>>>(x, e1w, e1b, h1);
    k2_enc2<<<2048, 256, 0, stream>>>(h1, e2w, e2b, tok);
    k3_vq  <<<1024, 256, K3_LDS_SIZE, stream>>>(tok, cbf, hq, idx_out, loss);
    k4_dec1<<<2048, 256, 0, stream>>>(hq, d1w, d1b, y1);
    k5_dec2<<<8192, 256, 0, stream>>>(y1, d2w, d2b, y);
}
